// Rel0SelfAttention_21680994910236
// MI455X (gfx1250) — compile-verified, matrix operand order checked by reading, not run on hardware
//
#include <hip/hip_runtime.h>
#include <hip/hip_bf16.h>
#include <math.h>

typedef __attribute__((ext_vector_type(16))) _Float16 v16h;
typedef __attribute__((ext_vector_type(8)))  float    v8f;

#define HID 1024
#define NH  16
#define DH  64
#define BB  2
#define SS  2048
#define LOG2E 1.4426950408889634f

// ---------------- prep kernels ----------------
__global__ void cvt_f16(const float* __restrict__ src, _Float16* __restrict__ dst, int n) {
  int i = blockIdx.x * blockDim.x + threadIdx.x;
  int stride = gridDim.x * blockDim.x;
  for (; i < n; i += stride) dst[i] = (_Float16)src[i];
}

// W is [in][out] (1024x1024); produce WT[out][in] in f16 so B-fragments are K-contiguous
__global__ void transpose_w(const float* __restrict__ W, _Float16* __restrict__ WT) {
  int i = blockIdx.x * blockDim.x + threadIdx.x;
  int stride = gridDim.x * blockDim.x;
  for (; i < HID * HID; i += stride) {
    int r = i >> 10, c = i & 1023;
    WT[(size_t)c * HID + r] = (_Float16)W[i];
  }
}

// ---------------- projection GEMM (WMMA f16 -> f32) ----------------
// Out[M,N] = X16[M,1024] @ WT16[N,1024]^T (+bias). One 16x16 tile per wave, K=1024.
// mode 0: q -> out0 = q+bq+rwb, out1 = q+bq+rrb  layout [b][n][s][d]
// mode 1: k -> out0 layout [b][n][s][d]
// mode 2: v -> out0 layout [b][n][d][s]   (transposed for PV B-fragments)
// mode 3: r_head_k -> out0 layout [n][s][d]
__global__ __launch_bounds__(256) void proj_gemm(
    const _Float16* __restrict__ X, const _Float16* __restrict__ WT,
    const float* __restrict__ bias,
    const float* __restrict__ add0, const float* __restrict__ add1,
    _Float16* __restrict__ out0, _Float16* __restrict__ out1,
    int Mtok, int mode) {
  const int lane = threadIdx.x & 31;
  const int wave = threadIdx.x >> 5;
  const int hsel = lane >> 4;   // which 16-wide K half this lane holds
  const int lr   = lane & 15;
  const int nTilesN = HID / 16;
  const int nTiles  = (Mtok / 16) * nTilesN;
  int tile = blockIdx.x * 8 + wave;
  if (tile >= nTiles) return;
  const int M0 = (tile / nTilesN) * 16;
  const int N0 = (tile % nTilesN) * 16;

  v8f acc = {};
  const _Float16* xrow = X  + (size_t)(M0 + lr) * HID + hsel * 16;
  const _Float16* wrow = WT + (size_t)(N0 + lr) * HID + hsel * 16;
#pragma unroll 4
  for (int kk = 0; kk < HID / 32; ++kk) {
    v16h a  = *(const v16h*)(xrow + kk * 32);
    v16h bm = *(const v16h*)(wrow + kk * 32);
    acc = __builtin_amdgcn_wmma_f32_16x16x32_f16(false, a, false, bm, (short)0, acc, false, false);
  }

  const int col = N0 + lr;
  const float bb = bias ? bias[col] : 0.f;
  const int n = col >> 6, d = col & 63;
#pragma unroll
  for (int r = 0; r < 8; ++r) {
    int row = M0 + r + 8 * hsel;        // token (or r-position) index
    float v = acc[r] + bb;
    if (mode == 0) {
      int bi = row / SS, si = row % SS;
      size_t o = (((size_t)bi * NH + n) * SS + si) * DH + d;
      out0[o] = (_Float16)(v + add0[col]);
      out1[o] = (_Float16)(v + add1[col]);
    } else if (mode == 1) {
      int bi = row / SS, si = row % SS;
      out0[(((size_t)bi * NH + n) * SS + si) * DH + d] = (_Float16)v;
    } else if (mode == 2) {
      int bi = row / SS, si = row % SS;
      out0[(((size_t)bi * NH + n) * DH + d) * SS + si] = (_Float16)v;
    } else {
      out0[((size_t)n * SS + row) * DH + d] = (_Float16)v;
    }
  }
}

// ---------------- fused rel-shift flash attention ----------------
// One WG = (b, n, 16-row i-block). 4 waves, each owns a 512-col j-chunk.
__global__ __launch_bounds__(128) void relattn_flash(
    const _Float16* __restrict__ qw16, const _Float16* __restrict__ qr16,
    const _Float16* __restrict__ k16,  const _Float16* __restrict__ vT16,
    const _Float16* __restrict__ rk16, float* __restrict__ out) {
  __shared__ float    band[17 * 2048];        // BDraw rows i0..i0+16   (136 KB)
  __shared__ _Float16 pb[4 * 16 * 48];        // P tiles, padded stride 48
  __shared__ float    cpart[4 * 16 * 64];     // per-wave ctx partials
  __shared__ float    mpart[4 * 16];
  __shared__ float    lpart[4 * 16];

  const int tid  = threadIdx.x;
  const int lane = tid & 31, wave = tid >> 5;
  const int hsel = lane >> 4, lr = lane & 15;
  const int nib = SS / 16;
  const int wg = blockIdx.x;
  const int ib = wg % nib;
  const int n  = (wg / nib) % NH;
  const int b  = wg / (nib * NH);
  const int i0 = ib * 16;
  const size_t bn = (size_t)b * NH + n;

  // ---- prelude: BDraw band (qr @ rk^T), each wave 512 cols ----
  {
    const _Float16* qrb = qr16 + (bn * SS + i0 + lr) * DH + hsel * 16;
    v16h aR0 = *(const v16h*)(qrb);
    v16h aR1 = *(const v16h*)(qrb + 32);
    for (int t = 0; t < 32; ++t) {
      int J = wave * 512 + t * 16;
      const _Float16* kb = rk16 + ((size_t)n * SS + J + lr) * DH + hsel * 16;
      v16h b0 = *(const v16h*)(kb);
      v16h b1 = *(const v16h*)(kb + 32);
      v8f acc = {};
      acc = __builtin_amdgcn_wmma_f32_16x16x32_f16(false, aR0, false, b0, (short)0, acc, false, false);
      acc = __builtin_amdgcn_wmma_f32_16x16x32_f16(false, aR1, false, b1, (short)0, acc, false, false);
#pragma unroll
      for (int r = 0; r < 8; ++r) band[(r + 8 * hsel) * 2048 + J + lr] = acc[r];
    }
  }
  // 17th row (only consumed by rel-shift branch j>i+1 of the last block row)
  if (i0 + 16 < SS) {
    const _Float16* qrow = qr16 + (bn * SS + i0 + 16) * DH;
    for (int jj = tid; jj < SS; jj += 128) {
      const _Float16* kr = rk16 + ((size_t)n * SS + jj) * DH;
      float s = 0.f;
#pragma unroll 8
      for (int d = 0; d < DH; ++d) s += (float)qrow[d] * (float)kr[d];
      band[16 * 2048 + jj] = s;
    }
  }
  __syncthreads();

  // ---- flash loop ----
  const _Float16* qwb = qw16 + (bn * SS + i0 + lr) * DH + hsel * 16;
  v16h aW0 = *(const v16h*)(qwb);
  v16h aW1 = *(const v16h*)(qwb + 32);

  v8f acc0 = {}, acc1 = {}, acc2 = {}, acc3 = {};
  float mrow[8], lrow[8];
#pragma unroll
  for (int r = 0; r < 8; ++r) { mrow[r] = -1e30f; lrow[r] = 0.f; }

  _Float16* pw = pb + wave * 16 * 48;

  for (int jb = wave * 512; jb < wave * 512 + 512; jb += 32) {
    // two 16-col score tiles (AC via WMMA)
    v8f s0 = {}, s1 = {};
    {
      const _Float16* kb = k16 + (bn * SS + jb + lr) * DH + hsel * 16;
      v16h b0 = *(const v16h*)(kb);
      v16h b1 = *(const v16h*)(kb + 32);
      s0 = __builtin_amdgcn_wmma_f32_16x16x32_f16(false, aW0, false, b0, (short)0, s0, false, false);
      s0 = __builtin_amdgcn_wmma_f32_16x16x32_f16(false, aW1, false, b1, (short)0, s0, false, false);
    }
    {
      const _Float16* kb = k16 + (bn * SS + jb + 16 + lr) * DH + hsel * 16;
      v16h b0 = *(const v16h*)(kb);
      v16h b1 = *(const v16h*)(kb + 32);
      s1 = __builtin_amdgcn_wmma_f32_16x16x32_f16(false, aW0, false, b0, (short)0, s1, false, false);
      s1 = __builtin_amdgcn_wmma_f32_16x16x32_f16(false, aW1, false, b1, (short)0, s1, false, false);
    }
    const int j0 = jb + lr, j1 = jb + 16 + lr;
#pragma unroll
    for (int r = 0; r < 8; ++r) {
      const int M = r + 8 * hsel;
      const int i = i0 + M;
      // rel-shift gather: BDshift[i,j] from LDS band
      float bd0, bd1;
      if (j0 <= i)          bd0 = band[M * 2048 + (SS - 1) + j0 - i];
      else if (j0 == i + 1) bd0 = 0.f;
      else                  bd0 = band[(M + 1) * 2048 + j0 - i - 2];
      if (j1 <= i)          bd1 = band[M * 2048 + (SS - 1) + j1 - i];
      else if (j1 == i + 1) bd1 = 0.f;
      else                  bd1 = band[(M + 1) * 2048 + j1 - i - 2];
      float t0 = (s0[r] + bd0) * 0.125f;
      float t1 = (s1[r] + bd1) * 0.125f;
      // row max over 32 cols: pairwise then width-16 shuffle tree (wave32)
      float mt = fmaxf(t0, t1);
      mt = fmaxf(mt, __shfl_xor(mt, 1, 16));
      mt = fmaxf(mt, __shfl_xor(mt, 2, 16));
      mt = fmaxf(mt, __shfl_xor(mt, 4, 16));
      mt = fmaxf(mt, __shfl_xor(mt, 8, 16));
      float newm = fmaxf(mrow[r], mt);
      float sc = exp2f((mrow[r] - newm) * LOG2E);
      float p0 = exp2f((t0 - newm) * LOG2E);
      float p1 = exp2f((t1 - newm) * LOG2E);
      float ps = p0 + p1;
      ps += __shfl_xor(ps, 1, 16);
      ps += __shfl_xor(ps, 2, 16);
      ps += __shfl_xor(ps, 4, 16);
      ps += __shfl_xor(ps, 8, 16);
      lrow[r] = lrow[r] * sc + ps;
      mrow[r] = newm;
      acc0[r] *= sc; acc1[r] *= sc; acc2[r] *= sc; acc3[r] *= sc;
      pw[M * 48 + lr]      = (_Float16)p0;
      pw[M * 48 + 16 + lr] = (_Float16)p1;
    }
    // P back as A-fragment (K=32), then PV: 4 WMMAs over d
    v16h aP = *(const v16h*)(pw + lr * 48 + hsel * 16);
    const _Float16* vb = vT16 + (bn * DH + lr) * SS + jb + hsel * 16;
    v16h bv0 = *(const v16h*)(vb);
    v16h bv1 = *(const v16h*)(vb + (size_t)16 * SS);
    v16h bv2 = *(const v16h*)(vb + (size_t)32 * SS);
    v16h bv3 = *(const v16h*)(vb + (size_t)48 * SS);
    acc0 = __builtin_amdgcn_wmma_f32_16x16x32_f16(false, aP, false, bv0, (short)0, acc0, false, false);
    acc1 = __builtin_amdgcn_wmma_f32_16x16x32_f16(false, aP, false, bv1, (short)0, acc1, false, false);
    acc2 = __builtin_amdgcn_wmma_f32_16x16x32_f16(false, aP, false, bv2, (short)0, acc2, false, false);
    acc3 = __builtin_amdgcn_wmma_f32_16x16x32_f16(false, aP, false, bv3, (short)0, acc3, false, false);
  }

  // ---- cross-wave combine ----
#pragma unroll
  for (int r = 0; r < 8; ++r) {
    const int M = r + 8 * hsel;
    cpart[(wave * 16 + M) * 64 + lr]      = acc0[r];
    cpart[(wave * 16 + M) * 64 + 16 + lr] = acc1[r];
    cpart[(wave * 16 + M) * 64 + 32 + lr] = acc2[r];
    cpart[(wave * 16 + M) * 64 + 48 + lr] = acc3[r];
    if (lr == 0) { mpart[wave * 16 + M] = mrow[r]; lpart[wave * 16 + M] = lrow[r]; }
  }
  __syncthreads();
  for (int idx = tid; idx < 16 * 64; idx += 128) {
    const int row = idx >> 6, d = idx & 63;
    float m = fmaxf(fmaxf(mpart[row], mpart[16 + row]), fmaxf(mpart[32 + row], mpart[48 + row]));
    float c = 0.f, l = 0.f;
#pragma unroll
    for (int w = 0; w < 4; ++w) {
      float e = exp2f((mpart[w * 16 + row] - m) * LOG2E);
      l += e * lpart[w * 16 + row];
      c += e * cpart[(w * 16 + row) * 64 + d];
    }
    out[((size_t)b * SS + i0 + row) * HID + n * DH + d] = c / l;
  }
}

// ---------------- host ----------------
extern "C" void kernel_launch(void* const* d_in, const int* in_sizes, int n_in,
                              void* d_out, int out_size, void* d_ws, size_t ws_size,
                              hipStream_t stream) {
  (void)in_sizes; (void)n_in; (void)out_size; (void)ws_size;
  const float* hs  = (const float*)d_in[0];
  const float* r   = (const float*)d_in[1];
  const float* rwb = (const float*)d_in[2];
  const float* rrb = (const float*)d_in[3];
  const float* Wq  = (const float*)d_in[4];
  const float* bq  = (const float*)d_in[5];
  const float* Wk  = (const float*)d_in[6];
  const float* bk  = (const float*)d_in[7];
  const float* Wv  = (const float*)d_in[8];
  const float* bv  = (const float*)d_in[9];
  const float* Wr  = (const float*)d_in[10];
  float* out = (float*)d_out;

  char* p = (char*)d_ws;
  auto alloc = [&](size_t bytes) { char* q = p; p += (bytes + 255) & ~(size_t)255; return q; };
  _Float16* hs16 = (_Float16*)alloc((size_t)BB * SS * HID * 2);
  _Float16* r16  = (_Float16*)alloc((size_t)SS * HID * 2);
  _Float16* WqT  = (_Float16*)alloc((size_t)HID * HID * 2);
  _Float16* WkT  = (_Float16*)alloc((size_t)HID * HID * 2);
  _Float16* WvT  = (_Float16*)alloc((size_t)HID * HID * 2);
  _Float16* WrT  = (_Float16*)alloc((size_t)HID * HID * 2);
  _Float16* qw16 = (_Float16*)alloc((size_t)BB * NH * SS * DH * 2);
  _Float16* qr16 = (_Float16*)alloc((size_t)BB * NH * SS * DH * 2);
  _Float16* k16  = (_Float16*)alloc((size_t)BB * NH * SS * DH * 2);
  _Float16* vT16 = (_Float16*)alloc((size_t)BB * NH * DH * SS * 2);
  _Float16* rk16 = (_Float16*)alloc((size_t)NH * SS * DH * 2);

  cvt_f16<<<1024, 256, 0, stream>>>(hs, hs16, BB * SS * HID);
  cvt_f16<<<512, 256, 0, stream>>>(r, r16, SS * HID);
  transpose_w<<<1024, 256, 0, stream>>>(Wq, WqT);
  transpose_w<<<1024, 256, 0, stream>>>(Wk, WkT);
  transpose_w<<<1024, 256, 0, stream>>>(Wv, WvT);
  transpose_w<<<1024, 256, 0, stream>>>(Wr, WrT);

  const int tilesQ = (BB * SS / 16) * (HID / 16);  // 16384
  const int tilesR = (SS / 16) * (HID / 16);       //  8192
  proj_gemm<<<(tilesQ + 7) / 8, 256, 0, stream>>>(hs16, WqT, bq, rwb, rrb, qw16, qr16, BB * SS, 0);
  proj_gemm<<<(tilesQ + 7) / 8, 256, 0, stream>>>(hs16, WkT, bk, nullptr, nullptr, k16, nullptr, BB * SS, 1);
  proj_gemm<<<(tilesQ + 7) / 8, 256, 0, stream>>>(hs16, WvT, bv, nullptr, nullptr, vT16, nullptr, BB * SS, 2);
  proj_gemm<<<(tilesR + 7) / 8, 256, 0, stream>>>(r16, WrT, nullptr, nullptr, nullptr, rk16, nullptr, SS, 3);

  relattn_flash<<<BB * NH * (SS / 16), 128, 0, stream>>>(qw16, qr16, k16, vT16, rk16, out);
}
